// MultiheadAttentionEx_83416854823365
// MI455X (gfx1250) — matrix operand order repaired and checked by reading, not compiled, not run on hardware
//
#include <hip/hip_runtime.h>

typedef __attribute__((ext_vector_type(16))) _Float16 v16h;
typedef __attribute__((ext_vector_type(8)))  _Float16 v8h;
typedef __attribute__((ext_vector_type(8)))  float    v8f;
typedef __attribute__((ext_vector_type(4)))  unsigned int u32x4;
typedef __attribute__((ext_vector_type(8)))  int      i32x8;
typedef __attribute__((ext_vector_type(4)))  int      i32x4;

#define DEV __device__ __forceinline__

#if __has_include(<hip/amd_detail/amd_gfx1250_TDM.h>)
#define MHA_TDM_6ARG 1
#else
#define MHA_TDM_6ARG 0
#endif

constexpr int B = 8, S = 1024, D = 1024, H = 16, DK = 64;
constexpr size_t NTOK   = (size_t)B * S;     // 8192 tokens
constexpr size_t SZ_ACT = NTOK * D;          // 8M halfs per activation tensor
constexpr size_t SZ_W   = (size_t)D * D;     // 1M halfs per weight matrix

// workspace layout, in units of _Float16 elements
constexpr size_t OFF_QH  = 0;
constexpr size_t OFF_KH  = OFF_QH  + SZ_ACT;
constexpr size_t OFF_VH  = OFF_KH  + SZ_ACT;
constexpr size_t OFF_WQ  = OFF_VH  + SZ_ACT;
constexpr size_t OFF_WK  = OFF_WQ  + SZ_W;
constexpr size_t OFF_WV  = OFF_WK  + SZ_W;
constexpr size_t OFF_WO  = OFF_WV  + SZ_W;
constexpr size_t OFF_Q1  = OFF_WO  + SZ_W;   // [B,H,S,DK]  f16
constexpr size_t OFF_K1  = OFF_Q1  + SZ_ACT; // [B,H,S,DK]  f16
constexpr size_t OFF_V1T = OFF_K1  + SZ_ACT; // [B,H,DK,S]  f16 (transposed)
constexpr size_t OFF_AO  = OFF_V1T + SZ_ACT; // [B,S,D]     f16 attention out

DEV v8f wmma_f16(v16h a, v16h b, v8f c) {
    return __builtin_amdgcn_wmma_f32_16x16x32_f16(
        /*neg_a=*/false, a, /*neg_b=*/false, b,
        /*c_mod=*/(short)0, c, /*reuse_a=*/false, /*reuse_b=*/false);
}

// Build A fragment (16x32 f16): lane L holds row M = L%16;
// halfs [0..15] = K in [sel, sel+16),       sel  = (L<16 ? 0 : 16)
DEV v16h load_afrag(const _Float16* rowbase, int hi) {
    return *(const v16h*)(rowbase + hi * 16);
}

// TDM: async DMA of a 2D tile (tileD1 lines x tileD0 contiguous elements, 2B
// elements) from global to LDS.  Descriptor per CDNA5 ISA.
// Issue once per wave (TDM ignores EXEC); tracked by TENSORcnt.
DEV void tdm_load_2d(unsigned ldsOff, const _Float16* g,
                     unsigned tileD0, unsigned tileD1,
                     unsigned tenD0, unsigned tenD1,
                     unsigned long stride0 /* data_size units */) {
    const unsigned long ga = (unsigned long)(uintptr_t)g;
    u32x4 g0;
    g0[0] = 1u;                                   // count=1, user descriptor
    g0[1] = ldsOff;                               // lds_addr (bytes)
    g0[2] = (unsigned)(ga & 0xffffffffu);         // global_addr[31:0]
    g0[3] = (unsigned)((ga >> 32) & 0x1ffffffu)   // global_addr[56:32]
            | (2u << 30);                         // type=2 ("image")
    i32x8 g1;
    g1[0] = (int)(1u << 16);                      // wg_mask=0, data_size=2B
    g1[1] = (int)((tenD0 & 0xffffu) << 16);       // abar=0 | tensor_dim0[15:0]
    g1[2] = (int)(((tenD0 >> 16) & 0xffffu) | ((tenD1 & 0xffffu) << 16));
    g1[3] = (int)(((tenD1 >> 16) & 0xffffu) | ((tileD0 & 0xffffu) << 16));
    g1[4] = (int)(tileD1 & 0xffffu);              // tile_dim1 | tile_dim2=0
    g1[5] = (int)(unsigned)(stride0 & 0xffffffffu);     // dim0_stride[31:0]
    g1[6] = (int)(unsigned)((stride0 >> 32) & 0xffffu); // dim0_stride[47:32]
    g1[7] = 0;                                    // dim1_stride (unused, 2D)
    i32x4 gz = {0, 0, 0, 0};
#if MHA_TDM_6ARG
    i32x8 gx = {0, 0, 0, 0, 0, 0, 0, 0};
    __builtin_amdgcn_tensor_load_to_lds(g0, g1, gz, gz, gx, 0);
#else
    __builtin_amdgcn_tensor_load_to_lds(g0, g1, gz, gz, 0);
#endif
}

DEV unsigned lds_off_of(const void* p) {   // LDS byte offset = flat addr[31:0]
    return (unsigned)(uintptr_t)p;
}

__global__ void mha_cast_f32_f16(const float* __restrict__ src,
                                 _Float16* __restrict__ dst, int n) {
    int i = blockIdx.x * blockDim.x + threadIdx.x;
    if (i < n) dst[i] = (_Float16)src[i];
}

// Y = X (NTOK x D, f16) @ W^T (W is D x D row-major f16) + bias
// TDM double-buffers the 128x32 A tile and 64x32 B tile into LDS.
// MODE 0: store f16 as [B,H,S,DK]   (Q and K projections)
// MODE 1: store f16 as [B,H,DK,S]   (V projection, transposed)
// MODE 2: store f32 as [NTOK, D]    (output projection -> d_out)
template <int MODE>
__global__ __launch_bounds__(128) void mha_gemm_xwt(
    const _Float16* __restrict__ X, const _Float16* __restrict__ W,
    const float* __restrict__ bias, void* __restrict__ dst) {
    __shared__ __align__(16) _Float16 Abuf[2][128 * 32];
    __shared__ __align__(16) _Float16 Bbuf[2][64 * 32];

    const int lane = threadIdx.x & 31;
    const int wave = threadIdx.x >> 5;
    const int hi   = lane >> 4;   // 0/1 half-wave
    const int lcol = lane & 15;
    const int rowBase = blockIdx.x * 128 + wave * 32;  // token rows
    const int colBase = blockIdx.y * 64;               // feature cols
    const int blkRow  = blockIdx.x * 128;

    v8f acc[2][4];
#pragma unroll
    for (int rt = 0; rt < 2; ++rt)
#pragma unroll
        for (int ct = 0; ct < 4; ++ct)
#pragma unroll
            for (int i = 0; i < 8; ++i) acc[rt][ct][i] = 0.f;

    constexpr int NITER = D / 32;
    if (wave == 0) {
        tdm_load_2d(lds_off_of(&Abuf[0][0]), X + (size_t)blkRow * D,
                    32, 128, (unsigned)D, (unsigned)NTOK, (unsigned long)D);
        tdm_load_2d(lds_off_of(&Bbuf[0][0]), W + (size_t)colBase * D,
                    32, 64, (unsigned)D, (unsigned)D, (unsigned long)D);
    }
    for (int it = 0; it < NITER; ++it) {
        const int buf = it & 1;
        if (wave == 0) {
            if (it + 1 < NITER) {
                const int kk2 = (it + 1) * 32;
                tdm_load_2d(lds_off_of(&Abuf[buf ^ 1][0]),
                            X + (size_t)blkRow * D + kk2,
                            32, 128, (unsigned)D, (unsigned)NTOK, (unsigned long)D);
                tdm_load_2d(lds_off_of(&Bbuf[buf ^ 1][0]),
                            W + (size_t)colBase * D + kk2,
                            32, 64, (unsigned)D, (unsigned)D, (unsigned long)D);
                __builtin_amdgcn_s_wait_tensorcnt(2);  // current buffer landed
            } else {
                __builtin_amdgcn_s_wait_tensorcnt(0);
            }
        }
        __syncthreads();  // publish buffer `buf`
        v16h a[2];
#pragma unroll
        for (int rt = 0; rt < 2; ++rt)
            a[rt] = load_afrag(&Abuf[buf][(wave * 32 + rt * 16 + lcol) * 32], hi);
        v16h bf[4];
#pragma unroll
        for (int ct = 0; ct < 4; ++ct)
            bf[ct] = *(const v16h*)(&Bbuf[buf][(ct * 16 + lcol) * 32 + hi * 16]);
#pragma unroll
        for (int rt = 0; rt < 2; ++rt)
#pragma unroll
            for (int ct = 0; ct < 4; ++ct)
                acc[rt][ct] = wmma_f16(a[rt], bf[ct], acc[rt][ct]);
        __syncthreads();  // all waves done reading before TDM overwrites
    }

#pragma unroll
    for (int ct = 0; ct < 4; ++ct) {
        const int m  = colBase + ct * 16 + lcol;  // output feature
        const float bv = bias[m];
#pragma unroll
        for (int rt = 0; rt < 2; ++rt) {
#pragma unroll
            for (int r = 0; r < 8; ++r) {
                const int row = rowBase + rt * 16 + r + hi * 8;  // token index
                const float y = acc[rt][ct][r] + bv;
                if (MODE == 2) {
                    ((float*)dst)[(size_t)row * D + m] = y;
                } else {
                    const int b_ = row >> 10, s_ = row & (S - 1);
                    const int h_ = m >> 6,   d_ = m & (DK - 1);
                    const size_t idx =
                        (MODE == 0) ? ((size_t)(b_ * H + h_) * S + s_) * DK + d_
                                    : ((size_t)(b_ * H + h_) * DK + d_) * S + s_;
                    ((_Float16*)dst)[idx] = (_Float16)y;
                }
            }
        }
    }
}

// Causal flash attention. One wave per (b, h, 16 query rows).
// Q,K: [B,H,S,DK] f16; Vt: [B,H,DK,S] f16; AO: [B,S,D] f16.
__global__ __launch_bounds__(32) void mha_flash_attn(
    const _Float16* __restrict__ Q, const _Float16* __restrict__ K,
    const _Float16* __restrict__ Vt, _Float16* __restrict__ AO) {
    __shared__ __align__(32) _Float16 Plds[16 * 32];
    const int lane = threadIdx.x;
    const int hi = lane >> 4, lcol = lane & 15;
    const int nqt = S / 16;
    const int qt = blockIdx.x % nqt;
    const int bh = blockIdx.x / nqt;
    const int b_ = bh / H, h_ = bh % H;
    const _Float16* Qp = Q  + (size_t)bh * S * DK;
    const _Float16* Kp = K  + (size_t)bh * S * DK;
    const _Float16* Vp = Vt + (size_t)bh * DK * S;

    v16h qa[2];
    {
        const int qrow = qt * 16 + lcol;
#pragma unroll
        for (int ks = 0; ks < 2; ++ks)
            qa[ks] = load_afrag(Qp + (size_t)qrow * DK + ks * 32, hi);
    }

    float mrow[8], lrow[8];
    v8f o[4];
#pragma unroll
    for (int r = 0; r < 8; ++r) { mrow[r] = -1e30f; lrow[r] = 0.f; }
#pragma unroll
    for (int ct = 0; ct < 4; ++ct)
#pragma unroll
        for (int i = 0; i < 8; ++i) o[ct][i] = 0.f;

    const int nkb = qt / 2 + 1;  // causal: key blocks with kBase <= qBase+15
    for (int kb = 0; kb < nkb; ++kb) {
        const int kBase = kb * 32;
        if (kb + 1 < nkb)  // hint next key block toward the caches
            __builtin_prefetch(Kp + (size_t)(kBase + 32) * DK, 0, 1);
        v8f st[2];
#pragma unroll
        for (int t = 0; t < 2; ++t) {
            const int ktok = kBase + t * 16 + lcol;  // B column = key token
            const v16h k0 = *(const v16h*)(Kp + (size_t)ktok * DK + hi * 16);
            const v16h k1 = *(const v16h*)(Kp + (size_t)ktok * DK + 32 + hi * 16);
            v8f z;
#pragma unroll
            for (int i = 0; i < 8; ++i) z[i] = 0.f;
            z = wmma_f16(qa[0], k0, z);
            st[t] = wmma_f16(qa[1], k1, z);
        }
#pragma unroll
        for (int r = 0; r < 8; ++r) {
            const int qg = qt * 16 + r + hi * 8;
#pragma unroll
            for (int t = 0; t < 2; ++t) {
                const int kg = kBase + t * 16 + lcol;
                float sv = st[t][r] * 0.125f;     // 1/sqrt(DK)
                if (kg > qg) sv = -1e30f;
                st[t][r] = sv;
            }
            float rm = fmaxf(st[0][r], st[1][r]);
#pragma unroll
            for (int msk = 1; msk <= 8; msk <<= 1)
                rm = fmaxf(rm, __shfl_xor(rm, msk, 32));
            const float nm = fmaxf(mrow[r], rm);
            const float alpha = __expf(mrow[r] - nm);
            mrow[r] = nm;
            const float p0 = __expf(st[0][r] - nm);
            const float p1 = __expf(st[1][r] - nm);
            st[0][r] = p0; st[1][r] = p1;
            float rs = p0 + p1;
#pragma unroll
            for (int msk = 1; msk <= 8; msk <<= 1)
                rs += __shfl_xor(rs, msk, 32);
            lrow[r] = lrow[r] * alpha + rs;
#pragma unroll
            for (int ct = 0; ct < 4; ++ct) o[ct][r] *= alpha;
        }
        __syncthreads();
#pragma unroll
        for (int r = 0; r < 8; ++r)
#pragma unroll
            for (int t = 0; t < 2; ++t)
                Plds[(r + hi * 8) * 32 + t * 16 + lcol] = (_Float16)st[t][r];
        __syncthreads();
        const v16h pa = load_afrag(&Plds[lcol * 32], hi);
#pragma unroll
        for (int ct = 0; ct < 4; ++ct) {
            const int dk = ct * 16 + lcol;  // B column = dk
            const v16h vb = *(const v16h*)(Vp + (size_t)dk * S + kBase + hi * 16);
            o[ct] = wmma_f16(pa, vb, o[ct]);
        }
    }
#pragma unroll
    for (int ct = 0; ct < 4; ++ct) {
        const int feat = h_ * DK + ct * 16 + lcol;
#pragma unroll
        for (int r = 0; r < 8; ++r) {
            const int tok = qt * 16 + r + hi * 8;
            const float y = o[ct][r] / lrow[r];
            AO[((size_t)(b_ * S + tok)) * D + feat] = (_Float16)y;
        }
    }
}

extern "C" void kernel_launch(void* const* d_in, const int* in_sizes, int n_in,
                              void* d_out, int out_size, void* d_ws, size_t ws_size,
                              hipStream_t stream) {
    const float* q  = (const float*)d_in[0];
    const float* k  = (const float*)d_in[1];
    const float* v  = (const float*)d_in[2];
    // d_in[3] = causal mask (int), recomputed analytically -> unused
    const float* Wq = (const float*)d_in[4];
    const float* bq = (const float*)d_in[5];
    const float* Wk = (const float*)d_in[6];
    const float* bk = (const float*)d_in[7];
    const float* Wv = (const float*)d_in[8];
    const float* bv = (const float*)d_in[9];
    const float* Wo = (const float*)d_in[10];
    const float* bo = (const float*)d_in[11];
    _Float16* ws = (_Float16*)d_ws;

    const int nAct = (int)SZ_ACT, nW = (int)SZ_W;
    const int cb = 256;
    mha_cast_f32_f16<<<(nAct + cb - 1) / cb, cb, 0, stream>>>(q,  ws + OFF_QH, nAct);
    mha_cast_f32_f16<<<(nAct + cb - 1) / cb, cb, 0, stream>>>(k,  ws + OFF_KH, nAct);
    mha_cast_f32_f16<<<(nAct + cb - 1) / cb, cb, 0, stream>>>(v,  ws + OFF_VH, nAct);
    mha_cast_f32_f16<<<(nW   + cb - 1) / cb, cb, 0, stream>>>(Wq, ws + OFF_WQ, nW);
    mha_cast_f32_f16<<<(nW   + cb - 1) / cb, cb, 0, stream>>>(Wk, ws + OFF_WK, nW);
    mha_cast_f32_f16<<<(nW   + cb - 1) / cb, cb, 0, stream>>>(Wv, ws + OFF_WV, nW);
    mha_cast_f32_f16<<<(nW   + cb - 1) / cb, cb, 0, stream>>>(Wo, ws + OFF_WO, nW);

    const dim3 gg((unsigned)(NTOK / 128), (unsigned)(D / 64));
    mha_gemm_xwt<0><<<gg, 128, 0, stream>>>(ws + OFF_QH, ws + OFF_WQ, bq, ws + OFF_Q1);
    mha_gemm_xwt<0><<<gg, 128, 0, stream>>>(ws + OFF_KH, ws + OFF_WK, bk, ws + OFF_K1);
    mha_gemm_xwt<1><<<gg, 128, 0, stream>>>(ws + OFF_VH, ws + OFF_WV, bv, ws + OFF_V1T);

    mha_flash_attn<<<B * H * (S / 16), 32, 0, stream>>>(
        ws + OFF_Q1, ws + OFF_K1, ws + OFF_V1T, ws + OFF_AO);

    mha_gemm_xwt<2><<<gg, 128, 0, stream>>>(ws + OFF_AO, ws + OFF_WO, bo, d_out);
}
